// AttentionDecoder_32641751449989
// MI455X (gfx1250) — matrix operand order repaired and checked by reading, not compiled, not run on hardware
//
#include <hip/hip_runtime.h>
#include <hip/hip_bf16.h>
#include <math.h>

typedef __attribute__((ext_vector_type(16))) _Float16 v16h;
typedef __attribute__((ext_vector_type(8)))  float    v8f;
typedef __attribute__((ext_vector_type(4)))  int      v4i;

// CDNA5 async global->LDS path (guarded: falls back to VGPR staging if the
// builtins are not declared by this toolchain). Probe round 2 confirmed the
// builtin exists with signature (global v4i*, lds v4i*, imm offset, imm cpol).
#if __has_builtin(__builtin_amdgcn_global_load_async_to_lds_b128) && \
    __has_builtin(__builtin_amdgcn_s_wait_asynccnt)
#define USE_ASYNC_LDS 1
typedef __attribute__((address_space(1))) v4i gv4i;
typedef __attribute__((address_space(3))) v4i lv4i;
#else
#define USE_ASYNC_LDS 0
#endif

// ---------- helpers ----------

// Monotonic float -> uint mapping (order-preserving for all finite + inf values).
__device__ __forceinline__ unsigned fkey(float f) {
  unsigned u = __float_as_uint(f);
  return (u & 0x80000000u) ? ~u : (u | 0x80000000u);
}
__device__ __forceinline__ float keyToFloat(unsigned k) {
  unsigned u = (k & 0x80000000u) ? (k & 0x7FFFFFFFu) : ~k;
  return __uint_as_float(u);
}

__device__ __forceinline__ unsigned rotl32(unsigned x, unsigned n) {
  return (x << n) | (x >> (32u - n));
}

// Threefry2x32-style counter RNG (key = (0,42) to mirror jax.random.key(42)).
__device__ __forceinline__ void threefry2x32(unsigned k0, unsigned k1,
                                             unsigned c0, unsigned c1,
                                             unsigned& o0, unsigned& o1) {
  const unsigned R[8] = {13u, 15u, 26u, 6u, 17u, 29u, 16u, 24u};
  unsigned ks[3] = {k0, k1, 0x1BD11BDAu ^ k0 ^ k1};
  unsigned x0 = c0 + ks[0];
  unsigned x1 = c1 + ks[1];
#pragma unroll
  for (int g = 0; g < 5; ++g) {
#pragma unroll
    for (int r = 0; r < 4; ++r) {
      x0 += x1;
      x1 = rotl32(x1, R[(g & 1) * 4 + r]);
      x1 ^= x0;
    }
    x0 += ks[(g + 1) % 3];
    x1 += ks[(g + 2) % 3] + (unsigned)(g + 1);
  }
  o0 = x0; o1 = x1;
}

// ---------- workspace layout (floats at d_ws) ----------
// [0..127]  : qk vector (query @ Wk)
// [128]     : maxKey   (unsigned, fkey of max masked score)
// [129]     : Z        (float, softmax denominator)
// [130]     : thrKey   (unsigned, fkey of 50th-largest masked score)
// [131]     : scale    (float, 1 / (Z * (S_top/Z + 1e-10)))
// [136..137]: pack     (u64, gumbel-argmax {fkey(val)<<32 | ~idx})

// ---------- kernel 1: query & qk, init slots ----------
__global__ __launch_bounds__(128) void setup_kernel(
    const float* __restrict__ cur, const float* __restrict__ ctx,
    const float* __restrict__ Wq, const float* __restrict__ Wk,
    float* __restrict__ qk, unsigned* __restrict__ maxKey,
    float* __restrict__ Zslot, unsigned long long* __restrict__ pack) {
  __shared__ float q[128];
  const int t = threadIdx.x;  // 0..127
  float acc = 0.0f;
  for (int j = 0; j < 128; ++j) acc += cur[j] * Wq[t * 256 + j];
  for (int j = 0; j < 128; ++j) acc += ctx[j] * Wq[t * 256 + 128 + j];
  q[t] = acc;
  __syncthreads();
  float a2 = 0.0f;
  for (int h = 0; h < 128; ++h) a2 += q[h] * Wk[h * 128 + t];
  qk[t] = a2;
  if (t == 0) { *maxKey = 0u; *Zslot = 0.0f; *pack = 0ull; }
}

// ---------- kernel 2: WMMA score tiles (the bandwidth-bound pass) ----------
// 8 waves / block, one 16-candidate tile per wave; tiles staged into LDS via
// CDNA5 async global->LDS DMA (ASYNCcnt) when available.
// scores[n] = mask[n] ? dot(qk, cand[n]) : -inf ; global max via fkey atomicMax.
__global__ __launch_bounds__(256) void scores_kernel(
    const float* __restrict__ cand, const float* __restrict__ qk,
    const int* __restrict__ mask, float* __restrict__ scores,
    unsigned* __restrict__ maxKeySlot, int nTiles, int N) {
  __shared__ float ldsF[8][16 * 128];  // 64 KB of the 320 KB WGP LDS (f32 tiles)
  const int wave = threadIdx.x >> 5;
  const int lane = threadIdx.x & 31;
  const int tile = blockIdx.x * 8 + wave;
  const int activeTile = (tile < nTiles) ? tile : 0;  // idle waves shadow tile 0

  const float4* src = (const float4*)(cand + (size_t)activeTile * 16 * 128);
  float* myLds = &ldsF[wave][0];

#if USE_ASYNC_LDS
  // Async DMA: memory -> LDS without VGPR round-trip; tracked with ASYNCcnt.
#pragma unroll
  for (int t = 0; t < 16; ++t) {
    const int idx = t * 32 + lane;  // float4 index 0..511, coalesced per wave
    __builtin_amdgcn_global_load_async_to_lds_b128(
        (gv4i*)(src + idx),
        (lv4i*)((float4*)myLds + idx),
        0, 0);
  }
  __builtin_amdgcn_s_wait_asynccnt(0);  // wave reads only its own tile
#else
#pragma unroll
  for (int t = 0; t < 16; ++t) {
    const int idx = t * 32 + lane;
    ((float4*)myLds)[idx] = src[idx];
  }
#endif
  __syncthreads();

  const int hi = lane >> 4;   // lane half (K split)
  const int m  = lane & 15;   // A-matrix row within tile
  v8f c = {};
#pragma unroll
  for (int kc = 0; kc < 4; ++kc) {     // E=128 -> 4 chunks of K=32
    v16h a, b;
    const float* rowF = &ldsF[wave][m * 128 + kc * 32];
    // A layout (ISA 7.12.2, 16-bit A 16x32): lanes 0-15 K={0..7,16..23}, lanes 16-31 +8.
#pragma unroll
    for (int j = 0; j < 16; ++j) {
      const int k = (j < 8) ? (j + 8 * hi) : (16 + (j - 8) + 8 * hi);
      a[j] = (_Float16)rowF[k];
    }
    // B layout (K x N, N across lanes): same K order as A: lanes 0-15 K={0..7,16..23}, lanes 16-31 +8.
    // Broadcast qk into every column so every D column holds the score vector.
    const float* qb = qk + kc * 32;
#pragma unroll
    for (int j = 0; j < 16; ++j) b[j] = (_Float16)qb[(j < 8) ? (j + 8 * hi) : (16 + (j - 8) + 8 * hi)];
    c = __builtin_amdgcn_wmma_f32_16x16x32_f16(false, a, false, b,
                                               (short)0, c, false, false);
  }

  // C/D layout: VGPR r -> row r (lanes 0-15) / r+8 (lanes 16-31). Columns identical.
  if (tile < nTiles && (lane == 0 || lane == 16)) {
    float localMax = -INFINITY;
#pragma unroll
    for (int r = 0; r < 8; ++r) {
      const int row = tile * 16 + hi * 8 + r;
      if (row < N) {
        const float s = (mask[row] != 0) ? c[r] : -INFINITY;
        scores[row] = s;
        localMax = fmaxf(localMax, s);
      }
    }
    atomicMax(maxKeySlot, fkey(localMax));
  }
}

// ---------- kernel 3: softmax denominator Z ----------
__global__ __launch_bounds__(256) void sumexp_kernel(
    const float* __restrict__ scores, const unsigned* __restrict__ maxKey,
    float* __restrict__ Zslot, int N) {
  __shared__ float bsum;
  if (threadIdx.x == 0) bsum = 0.0f;
  __syncthreads();
  const int i = blockIdx.x * 256 + threadIdx.x;
  float e = 0.0f;
  if (i < N) {
    const float M = keyToFloat(*maxKey);
    e = expf(scores[i] - M);  // -inf -> 0
  }
  atomicAdd(&bsum, e);
  __syncthreads();
  if (threadIdx.x == 0) atomicAdd(Zslot, bsum);
}

// ---------- kernel 4: top-K threshold via binary search on float bit-keys ----------
__global__ __launch_bounds__(1024) void select_kernel(
    const float* __restrict__ scores, const unsigned* __restrict__ maxKey,
    const float* __restrict__ Zslot, unsigned* __restrict__ thrSlot,
    float* __restrict__ scaleSlot, int N, int K) {
  __shared__ int cnt;
  __shared__ float fred;
  unsigned lo = 0u, hi = 0xFFFFFFFFu;
  for (int it = 0; it < 33 && lo < hi; ++it) {
    const unsigned mid =
        (unsigned)(((unsigned long long)lo + (unsigned long long)hi + 1ull) >> 1);
    if (threadIdx.x == 0) cnt = 0;
    __syncthreads();
    int local = 0;
    for (int i = threadIdx.x; i < N; i += 1024)
      if (fkey(scores[i]) >= mid) ++local;
    atomicAdd(&cnt, local);
    __syncthreads();
    const int c = cnt;
    __syncthreads();  // protect cnt from next-iteration reset
    if (c >= K) lo = mid; else hi = mid - 1u;
  }
  // lo == fkey(K-th largest masked score); count(>= lo) >= K (ties included, matches >=).
  const float M = keyToFloat(*maxKey);
  if (threadIdx.x == 0) fred = 0.0f;
  __syncthreads();
  float ls = 0.0f;
  for (int i = threadIdx.x; i < N; i += 1024) {
    const float s = scores[i];
    if (fkey(s) >= lo) ls += expf(s - M);
  }
  atomicAdd(&fred, ls);
  __syncthreads();
  if (threadIdx.x == 0) {
    const float Z = *Zslot;
    const float Sf = fred / Z;           // sum of filtered probs
    const float den = Sf + 1e-10f;       // reference's +1e-10
    *thrSlot = lo;
    *scaleSlot = 1.0f / (Z * den);       // filtered_norm_i = exp(s_i-M) * scale
  }
}

// ---------- kernel 5a: Gumbel-max categorical sample ----------
__global__ __launch_bounds__(256) void sample_kernel(
    const float* __restrict__ scores, const unsigned* __restrict__ maxKey,
    const unsigned* __restrict__ thrSlot, const float* __restrict__ scaleSlot,
    unsigned long long* __restrict__ pack, int N) {
  const int i = blockIdx.x * 256 + threadIdx.x;
  if (i >= N) return;
  const float M = keyToFloat(*maxKey);
  const unsigned thr = *thrSlot;
  const float scale = *scaleSlot;
  const float s = scores[i];
  float total = -INFINITY;
  if (fkey(s) >= thr) {
    const float fn = expf(s - M) * scale;
    const float logit = logf(fn);
    unsigned x0, x1;
    threefry2x32(0u, 42u, 0u, (unsigned)i, x0, x1);
    float u = (float)(x0 >> 8) * (1.0f / 16777216.0f);
    u = fminf(fmaxf(u, 1e-12f), 0.99999994f);
    const float g = -logf(-logf(u));
    total = logit + g;
  }
  const unsigned long long p =
      ((unsigned long long)fkey(total) << 32) |
      (unsigned long long)(0xFFFFFFFFu - (unsigned)i);  // ties -> smallest idx
  atomicMax(pack, p);
}

// ---------- kernel 5b: finalize log-probs in place + tail outputs ----------
__global__ __launch_bounds__(256) void finalize_kernel(
    float* __restrict__ out, const unsigned* __restrict__ maxKey,
    const unsigned* __restrict__ thrSlot, const float* __restrict__ scaleSlot,
    const unsigned long long* __restrict__ pack, int N) {
  const int i = blockIdx.x * 256 + threadIdx.x;
  if (i >= N) return;
  const float M = keyToFloat(*maxKey);
  const unsigned thr = *thrSlot;
  const float scale = *scaleSlot;
  const unsigned act = 0xFFFFFFFFu - (unsigned)((*pack) & 0xFFFFFFFFull);
  const float s = out[i];                      // staged score
  const bool sel = (fkey(s) >= thr);
  const float fn = sel ? expf(s - M) * scale : 0.0f;
  out[i] = logf(fn + 1e-10f);                  // log_probs_all
  if ((unsigned)i == act) {
    out[N]     = sel ? logf(fn) : -INFINITY;   // log_prob_action
    out[N + 1] = (float)i;                     // action_idx
  }
}

// ---------- launch ----------
extern "C" void kernel_launch(void* const* d_in, const int* in_sizes, int n_in,
                              void* d_out, int out_size, void* d_ws, size_t ws_size,
                              hipStream_t stream) {
  const float* cur  = (const float*)d_in[0];
  const float* ctx  = (const float*)d_in[1];
  const float* cand = (const float*)d_in[2];
  const float* Wq   = (const float*)d_in[3];
  const float* Wk   = (const float*)d_in[4];
  const int*   mask = (const int*)d_in[5];
  float* out = (float*)d_out;

  const int N = in_sizes[5];  // mask has N elements
  int Ktop = N / 2; if (Ktop < 1) Ktop = 1; if (Ktop > 50) Ktop = 50;

  float* wsf = (float*)d_ws;
  float*              qk       = wsf;
  unsigned*           maxKey   = (unsigned*)(wsf + 128);
  float*              Zslot    = wsf + 129;
  unsigned*           thrSlot  = (unsigned*)(wsf + 130);
  float*              scaleSlot= wsf + 131;
  unsigned long long* pack     = (unsigned long long*)(wsf + 136);  // 8B aligned

  setup_kernel<<<1, 128, 0, stream>>>(cur, ctx, Wq, Wk, qk, maxKey, Zslot, pack);

  const int nTiles  = (N + 15) / 16;           // 12500 for N=200000
  const int blocks2 = (nTiles + 7) / 8;
  scores_kernel<<<blocks2, 256, 0, stream>>>(cand, qk, mask, out, maxKey, nTiles, N);

  const int blocksN = (N + 255) / 256;
  sumexp_kernel<<<blocksN, 256, 0, stream>>>(out, maxKey, Zslot, N);

  select_kernel<<<1, 1024, 0, stream>>>(out, maxKey, Zslot, thrSlot, scaleSlot, N, Ktop);

  sample_kernel<<<blocksN, 256, 0, stream>>>(out, maxKey, thrSlot, scaleSlot, pack, N);

  finalize_kernel<<<blocksN, 256, 0, stream>>>(out, maxKey, thrSlot, scaleSlot, pack, N);
}
